// PointNetFeaturePropagation_53412213293940
// MI455X (gfx1250) — matrix operand order repaired and checked by reading, not compiled, not run on hardware
//
#include <hip/hip_runtime.h>

typedef __attribute__((ext_vector_type(16))) __bf16 bf16x16;
typedef __attribute__((ext_vector_type(8)))  __bf16 bf16x8;
typedef __attribute__((ext_vector_type(8)))  float  v8f;

#define B_    8
#define N_    8192
#define M_    2048
#define C1_   128
#define C2_   256
#define INCH_ 384
#define OUTC_ 256
#define ROWS_ (B_ * N_)          // 65536
#define BN_EPS_ 1e-5f

// ---------------------------------------------------------------------------
// 0) zero the BN-stats scratch (sum1[256] sq1[256] sum2[256] sq2[256])
// ---------------------------------------------------------------------------
__global__ void zero_stats_kernel(float* __restrict__ stats) {
  stats[threadIdx.x] = 0.0f;     // launched with 1 block x 1024 threads
}

// ---------------------------------------------------------------------------
// 1) fp32 -> bf16 weight conversion (w0: 256x384, w1: 256x256)
// ---------------------------------------------------------------------------
__global__ void convert_weights_kernel(const float* __restrict__ w0,
                                       const float* __restrict__ w1,
                                       __bf16* __restrict__ w0b,
                                       __bf16* __restrict__ w1b) {
  int i = blockIdx.x * blockDim.x + threadIdx.x;
  const int n0 = OUTC_ * INCH_;
  const int n1 = OUTC_ * OUTC_;
  if (i < n0)            w0b[i]      = (__bf16)w0[i];
  else if (i < n0 + n1)  w1b[i - n0] = (__bf16)w1[i - n0];
}

// ---------------------------------------------------------------------------
// 2) fused 3-NN + inverse-distance interpolation + concat -> x0 bf16 [ROWS,384]
//    grid: B * (N/256) blocks of 256 threads; xyz2[b] staged in LDS.
//    Faithful to the reference's flattened-gather bug: indices address
//    points2 batch 0 rows only.
// ---------------------------------------------------------------------------
__global__ void __launch_bounds__(256)
interp_concat_kernel(const float* __restrict__ xyz1,
                     const float* __restrict__ xyz2,
                     const float* __restrict__ points1,
                     const float* __restrict__ points2,
                     __bf16* __restrict__ X0) {
  __shared__ float sx[M_], sy[M_], sz[M_];     // 24 KB
  __shared__ int   sidx[256 * 3];
  __shared__ float sw[256 * 3];

  const int tid = threadIdx.x;
  const int b   = blockIdx.x >> 5;             // 32 blocks per batch (8192/256)
  const int n0  = (blockIdx.x & 31) * 256;

  // stage xyz2[b] into LDS
  for (int i = tid; i < M_; i += 256) {
    const float* p = xyz2 + ((size_t)b * M_ + i) * 3;
    sx[i] = p[0]; sy[i] = p[1]; sz[i] = p[2];
  }
  __syncthreads();

  // per-thread point: branchless top-3 nearest over all M
  const int n = n0 + tid;
  const float* p1 = xyz1 + ((size_t)b * N_ + n) * 3;
  const float px = p1[0], py = p1[1], pz = p1[2];

  float d0 = 3.0e38f, d1 = 3.0e38f, d2 = 3.0e38f;
  int   i0 = 0, i1 = 0, i2 = 0;
  for (int m = 0; m < M_; ++m) {
    const float dx = px - sx[m], dy = py - sy[m], dz = pz - sz[m];
    const float d  = dx * dx + dy * dy + dz * dz;
    const bool c0 = d < d0, c1 = d < d1, c2 = d < d2;
    d2 = c1 ? d1 : (c2 ? d : d2);  i2 = c1 ? i1 : (c2 ? m : i2);
    d1 = c0 ? d0 : (c1 ? d : d1);  i1 = c0 ? i0 : (c1 ? m : i1);
    d0 = c0 ? d  : d0;             i0 = c0 ? m  : i0;
  }
  float w0 = 1.0f / (d0 + 1e-8f);
  float w1 = 1.0f / (d1 + 1e-8f);
  float w2 = 1.0f / (d2 + 1e-8f);
  const float wsum = w0 + w1 + w2;
  w0 /= wsum; w1 /= wsum; w2 /= wsum;
  sidx[tid * 3 + 0] = i0; sidx[tid * 3 + 1] = i1; sidx[tid * 3 + 2] = i2;
  sw[tid * 3 + 0] = w0;   sw[tid * 3 + 1] = w1;   sw[tid * 3 + 2] = w2;
  __syncthreads();

  // cooperative coalesced gather: thread = channel c, loop points of block
  const int c = tid;                            // 0..255 == C2
  for (int p = 0; p < 256; ++p) {
    const int   j0 = sidx[p * 3 + 0], j1 = sidx[p * 3 + 1], j2 = sidx[p * 3 + 2];
    const float a0 = sw[p * 3 + 0],   a1 = sw[p * 3 + 1],   a2 = sw[p * 3 + 2];
    const float v  = a0 * points2[(size_t)j0 * C2_ + c]
                   + a1 * points2[(size_t)j1 * C2_ + c]
                   + a2 * points2[(size_t)j2 * C2_ + c];
    const size_t row = (size_t)b * N_ + n0 + p;
    X0[row * INCH_ + C1_ + c] = (__bf16)v;
  }

  // copy points1 into the first 128 channels
  for (int e = tid; e < 256 * C1_; e += 256) {
    const int p = e >> 7, ch = e & 127;
    const size_t row = (size_t)b * N_ + n0 + p;
    X0[row * INCH_ + ch] = (__bf16)points1[row * C1_ + ch];
  }
}

// ---------------------------------------------------------------------------
// 3) bf16 WMMA GEMM: Y[rows, OUTC] = X[rows, K] * W[OUTC, K]^T + bias
//    Each wave: 16 rows x 64 cols (A-frag reused across 4 WMMAs per K-chunk).
//    Block: 8 waves -> 128 rows x 64 cols.  grid = (rows/128, OUTC/64).
//    Epilogue: per-channel sum & sum-of-squares via LDS ds_add_f32 then one
//    global_atomic_add_f32 per channel per block (BN training stats).
// ---------------------------------------------------------------------------
template <bool STORE_F32>
__global__ void __launch_bounds__(256)
gemm_bn_kernel(const __bf16* __restrict__ X,
               const __bf16* __restrict__ W,
               const float*  __restrict__ bias,
               void* __restrict__ Y,
               float* __restrict__ statSum,
               float* __restrict__ statSq,
               int K) {
  __shared__ float lsum[64];
  __shared__ float lsq[64];
  const int tid = threadIdx.x;
  if (tid < 64) { lsum[tid] = 0.0f; lsq[tid] = 0.0f; }
  __syncthreads();

  const int wave = tid >> 5;
  const int lane = tid & 31;
  const int half = lane >> 4;        // 0: lanes 0-15, 1: lanes 16-31
  const int lm   = lane & 15;
  const int rowBase = blockIdx.x * 128 + wave * 16;
  const int colBase = blockIdx.y * 64;

  const __bf16* xrow = X + (size_t)(rowBase + lm) * K;

  v8f acc[4];
#pragma unroll
  for (int t = 0; t < 4; ++t) acc[t] = (v8f){};

  for (int k0 = 0; k0 < K; k0 += 32) {
    // A fragment (16x32): one contiguous 32-byte chunk per lane (k = 16*half + s)
    union { bf16x16 v; bf16x8 h[2]; } au;
    au.v = *(const bf16x16*)(xrow + k0 + half * 16);
#pragma unroll
    for (int t = 0; t < 4; ++t) {
      const int col = colBase + t * 16 + lm;
      // B fragment (32x16): B[k][n] = W[col][k]; one contiguous 32-byte chunk
      const bf16x16 bv = *(const bf16x16*)(W + (size_t)col * K + k0 + half * 16);
      acc[t] = __builtin_amdgcn_wmma_f32_16x16x32_bf16(
          false, au.v, false, bv, (short)0, acc[t], false, false);
    }
  }

  // epilogue: bias, store, BN statistics
#pragma unroll
  for (int t = 0; t < 4; ++t) {
    const int col = colBase + t * 16 + lm;
    const float bv = bias[col];
    float s = 0.0f, q = 0.0f;
#pragma unroll
    for (int j = 0; j < 8; ++j) {
      const float v = acc[t][j] + bv;
      s += v; q += v * v;
      const size_t row = (size_t)(rowBase + half * 8 + j);
      if (STORE_F32) ((float*)Y)[row * OUTC_ + col] = v;
      else           ((__bf16*)Y)[row * OUTC_ + col] = (__bf16)v;
    }
    atomicAdd(&lsum[t * 16 + lm], s);
    atomicAdd(&lsq[t * 16 + lm], q);
  }
  __syncthreads();
  if (tid < 64) {
    atomicAdd(&statSum[colBase + tid], lsum[tid]);
    atomicAdd(&statSq[colBase + tid],  lsq[tid]);
  }
}

// ---------------------------------------------------------------------------
// 4) BatchNorm(train stats) + ReLU, elementwise in place
// ---------------------------------------------------------------------------
__global__ void __launch_bounds__(256)
bn_relu_bf16_kernel(__bf16* __restrict__ Yio,
                    const float* __restrict__ sum, const float* __restrict__ sq,
                    const float* __restrict__ gamma, const float* __restrict__ beta,
                    float invN) {
  const size_t i = (size_t)blockIdx.x * 256 + threadIdx.x;
  const int c = (int)(i & (OUTC_ - 1));
  const float mu  = sum[c] * invN;
  const float var = sq[c] * invN - mu * mu;
  const float sc  = gamma[c] * rsqrtf(var + BN_EPS_);
  const float sh  = beta[c] - mu * sc;
  const float v   = (float)Yio[i] * sc + sh;
  Yio[i] = (__bf16)(v > 0.0f ? v : 0.0f);
}

__global__ void __launch_bounds__(256)
bn_relu_f32_kernel(float* __restrict__ Yio,
                   const float* __restrict__ sum, const float* __restrict__ sq,
                   const float* __restrict__ gamma, const float* __restrict__ beta,
                   float invN) {
  const size_t i = (size_t)blockIdx.x * 256 + threadIdx.x;
  const int c = (int)(i & (OUTC_ - 1));
  const float mu  = sum[c] * invN;
  const float var = sq[c] * invN - mu * mu;
  const float sc  = gamma[c] * rsqrtf(var + BN_EPS_);
  const float sh  = beta[c] - mu * sc;
  const float v   = Yio[i] * sc + sh;
  Yio[i] = v > 0.0f ? v : 0.0f;
}

// ---------------------------------------------------------------------------
// launch
// ---------------------------------------------------------------------------
extern "C" void kernel_launch(void* const* d_in, const int* in_sizes, int n_in,
                              void* d_out, int out_size, void* d_ws, size_t ws_size,
                              hipStream_t stream) {
  const float* xyz1    = (const float*)d_in[0];
  const float* xyz2    = (const float*)d_in[1];
  const float* points1 = (const float*)d_in[2];
  const float* points2 = (const float*)d_in[3];
  const float* w0 = (const float*)d_in[4];
  const float* b0 = (const float*)d_in[5];
  const float* g0 = (const float*)d_in[6];
  const float* be0 = (const float*)d_in[7];
  const float* w1 = (const float*)d_in[8];
  const float* b1 = (const float*)d_in[9];
  const float* g1 = (const float*)d_in[10];
  const float* be1 = (const float*)d_in[11];

  // workspace layout (aligned):
  //   [0,4096)                 : stats (sum1,sq1,sum2,sq2 : 4*256 f32)
  //   [4096, +196608)          : w0 bf16  (256*384)
  //   [200704, +131072)        : w1 bf16  (256*256)
  //   [331776, +50331648)      : x0 bf16  (65536*384)
  //   [50663424, +33554432)    : y1 bf16  (65536*256)
  char* ws = (char*)d_ws;
  float*  stats = (float*)ws;
  float*  sum1 = stats;       float* sq1 = stats + 256;
  float*  sum2 = stats + 512; float* sq2 = stats + 768;
  __bf16* w0b = (__bf16*)(ws + 4096);
  __bf16* w1b = (__bf16*)(ws + 200704);
  __bf16* x0  = (__bf16*)(ws + 331776);
  __bf16* y1  = (__bf16*)(ws + 50663424ull);
  float*  out = (float*)d_out;

  const float invN = 1.0f / (float)ROWS_;

  zero_stats_kernel<<<1, 1024, 0, stream>>>(stats);

  {
    const int total = OUTC_ * INCH_ + OUTC_ * OUTC_;
    convert_weights_kernel<<<(total + 255) / 256, 256, 0, stream>>>(w0, w1, w0b, w1b);
  }

  interp_concat_kernel<<<B_ * (N_ / 256), 256, 0, stream>>>(xyz1, xyz2, points1,
                                                            points2, x0);

  // layer 1: [65536 x 384] * [384 x 256] -> y1 (bf16) + stats
  gemm_bn_kernel<false><<<dim3(ROWS_ / 128, OUTC_ / 64), 256, 0, stream>>>(
      x0, w0b, b0, (void*)y1, sum1, sq1, INCH_);
  bn_relu_bf16_kernel<<<(ROWS_ * OUTC_) / 256, 256, 0, stream>>>(y1, sum1, sq1,
                                                                 g0, be0, invN);

  // layer 2: [65536 x 256] * [256 x 256] -> d_out (f32) + stats
  gemm_bn_kernel<true><<<dim3(ROWS_ / 128, OUTC_ / 64), 256, 0, stream>>>(
      y1, w1b, b1, (void*)out, sum2, sq2, OUTC_);
  bn_relu_f32_kernel<<<(ROWS_ * OUTC_) / 256, 256, 0, stream>>>(out, sum2, sq2,
                                                                g1, be1, invN);
}
